// TransformerMoE_49701361549374
// MI455X (gfx1250) — matrix operand order repaired and checked by reading, not compiled, not run on hardware
//
#include <hip/hip_runtime.h>

typedef __attribute__((ext_vector_type(16))) _Float16 v16h;
typedef __attribute__((ext_vector_type(8)))  _Float16 h8;
typedef __attribute__((ext_vector_type(8)))  float    v8f;

static constexpr int L  = 16, D = 256, F = 512, H = 8, E = 12;
static constexpr int B  = 4,  S = 512;
static constexpr int HD = D / H;          // 32
static constexpr int NT = B * S;          // 2048 tokens
static constexpr float EPS = 1e-5f;

// ---------------- WMMA fragment helpers (CDNA5 16x16x32 f16, wave32) ----------------
// A (16xK, row-major f16): lane holds row m=lane&15; K = k0 + 16*hi + j (16 contiguous f16).
__device__ __forceinline__ v16h load_a_f16(const _Float16* A, int row0, int k0, int lda, int lane) {
  const int m  = lane & 15;
  const int hi = (lane >> 4) & 1;
  const _Float16* p = A + (size_t)(row0 + m) * lda + k0 + hi * 16;
  h8 lo = *(const h8*)p;
  h8 hh = *(const h8*)(p + 8);
  return __builtin_shufflevector(lo, hh, 0,1,2,3,4,5,6,7,8,9,10,11,12,13,14,15);
}
// B (Kx16) sourced from a TRANSPOSED f16 matrix BT[N,K]: lane holds col n=lane&15;
// K = k0 + 16*hi + j  -> 16 contiguous f16 = two 16B loads.
__device__ __forceinline__ v16h load_bT_f16(const _Float16* BT, int n0, int k0, int ldk, int lane) {
  const int n  = lane & 15;
  const int hi = (lane >> 4) & 1;
  const _Float16* p = BT + (size_t)(n0 + n) * ldk + k0 + hi * 16;
  h8 lo = *(const h8*)p;
  h8 hh = *(const h8*)(p + 8);
  return __builtin_shufflevector(lo, hh, 0,1,2,3,4,5,6,7,8,9,10,11,12,13,14,15);
}
__device__ __forceinline__ v8f wmma32(v16h a, v16h b, v8f c) {
  return __builtin_amdgcn_wmma_f32_16x16x32_f16(false, a, false, b, (short)0, c, false, false);
}

// ---------------- kernels ----------------
__global__ void embed_kernel(const int* __restrict__ tok, const float* __restrict__ emb,
                             float* __restrict__ x, _Float16* __restrict__ xh) {
  const size_t i = (size_t)blockIdx.x * blockDim.x + threadIdx.x;   // NT*D threads
  const int t = (int)(i / D), d = (int)(i % D);
  const float v = emb[(size_t)tok[t] * D + d];
  x[i] = v;
  xh[i] = (_Float16)v;
}

// f32 [K,N] -> f16 transposed [N,K]; grid.z iterates experts with the given strides.
__global__ void convT_kernel(const float* __restrict__ in, _Float16* __restrict__ out,
                             int K, int N, size_t inStride, size_t outStride) {
  in  += (size_t)blockIdx.z * inStride;
  out += (size_t)blockIdx.z * outStride;
  __shared__ float t[32][33];
  const int kt = blockIdx.x << 5, nt = blockIdx.y << 5;
  const int tx = threadIdx.x & 31, ty = threadIdx.x >> 5;      // 256 threads = 8 row-groups
  for (int r = ty; r < 32; r += 8)
    t[r][tx] = in[(size_t)(kt + r) * N + nt + tx];
  __syncthreads();
  for (int r = ty; r < 32; r += 8)
    out[(size_t)(nt + r) * K + kt + tx] = (_Float16)t[tx][r];
}

// QKV: out = xh[NT,D] @ WT[D,D]^T + bias. One wave computes a 16x64 panel (A reused 4x).
// mode 0: f16 [NT,D]; mode 1: f16 [B,D,S] (v stored transposed)
__global__ void gemm_qkv_kernel(const _Float16* __restrict__ xh, const _Float16* __restrict__ WT,
                                const float* __restrict__ bias, _Float16* __restrict__ out, int mode) {
  const int lane = threadIdx.x & 31;
  const int wave = threadIdx.x >> 5;
  const int wid  = blockIdx.x * (blockDim.x >> 5) + wave;      // (NT/16)*(D/64) = 512 waves
  const int tm   = (wid / (D / 64)) << 4;
  const int tn0  = (wid % (D / 64)) << 6;
  v8f acc[4] = {v8f{}, v8f{}, v8f{}, v8f{}};
  for (int k = 0; k < D; k += 32) {
    v16h a = load_a_f16(xh, tm, k, D, lane);
#pragma unroll
    for (int t = 0; t < 4; ++t)
      acc[t] = wmma32(a, load_bT_f16(WT, tn0 + t * 16, k, D, lane), acc[t]);
  }
  const int n = lane & 15, hi = (lane >> 4) & 1;
#pragma unroll
  for (int t = 0; t < 4; ++t) {
    const int tn = tn0 + t * 16;
    const float bv = bias[tn + n];
    if (mode == 0) {
#pragma unroll
      for (int r = 0; r < 8; ++r)
        out[(size_t)(tm + r + hi * 8) * D + tn + n] = (_Float16)(acc[t][r] + bv);
    } else {
#pragma unroll
      for (int r = 0; r < 8; ++r) {
        const int rg = tm + r + hi * 8;               // token; 16-row tile never crosses batch
        const int bb = rg >> 9, s = rg & (S - 1);
        out[(size_t)bb * D * S + (size_t)(tn + n) * S + s] = (_Float16)(acc[t][r] + bv);
      }
    }
  }
}

// att[b,h,s,t] = sum_w q[...] k[...]. One wave per (bh, row-block): A fragment loaded once,
// reused across the whole causal row of N-tiles (nt <= mt).
__global__ void scores_kernel(const _Float16* __restrict__ qh, const _Float16* __restrict__ kh,
                              float* __restrict__ att) {
  const int lane = threadIdx.x & 31;
  const int wave = threadIdx.x >> 5;
  const int wid  = blockIdx.x * (blockDim.x >> 5) + wave;      // 32(bh)*32(mt) = 1024 waves
  const int mt = wid & 31, bh = wid >> 5;
  const int b = bh >> 3, h = bh & 7;
  const int sm = mt << 4;
  const _Float16* qb = qh + (size_t)b * S * D + h * HD;
  const _Float16* kb = kh + (size_t)b * S * D + h * HD;
  v16h a = load_a_f16(qb, sm, 0, D, lane);                     // K = HD = 32: one step
  float* ob = att + (size_t)bh * S * S;
  const int n = lane & 15, hi = (lane >> 4) & 1;
  for (int nt = 0; nt <= mt; ++nt) {                           // causal row of tiles
    v16h bb = load_bT_f16(kb, nt << 4, 0, D, lane);
    v8f acc = {};
    acc = wmma32(a, bb, acc);
#pragma unroll
    for (int r = 0; r < 8; ++r)
      ob[(size_t)(sm + r + hi * 8) * S + (nt << 4) + n] = acc[r];
  }
}

// causal softmax (f32 in), writes normalized probs as f16 (zeros past the diagonal)
__global__ void softmax_kernel(const float* __restrict__ att, _Float16* __restrict__ atth) {
  const int row = blockIdx.x;            // (b*H+h)*S + s
  const int s = row & (S - 1);
  const float* p = att + (size_t)row * S;
  _Float16* ph = atth + (size_t)row * S;
  __shared__ float red[256];
  const int tid = threadIdx.x;
  const int j0 = tid, j1 = tid + 256;
  const float v0 = (j0 <= s) ? p[j0] : -INFINITY;
  const float v1 = (j1 <= s) ? p[j1] : -INFINITY;
  red[tid] = fmaxf(v0, v1); __syncthreads();
  for (int off = 128; off > 0; off >>= 1) { if (tid < off) red[tid] = fmaxf(red[tid], red[tid + off]); __syncthreads(); }
  const float mx = red[0]; __syncthreads();
  const float e0 = (j0 <= s) ? __expf(v0 - mx) : 0.f;
  const float e1 = (j1 <= s) ? __expf(v1 - mx) : 0.f;
  red[tid] = e0 + e1; __syncthreads();
  for (int off = 128; off > 0; off >>= 1) { if (tid < off) red[tid] += red[tid + off]; __syncthreads(); }
  const float inv = 1.f / red[0];
  ph[j0] = (_Float16)(e0 * inv);
  ph[j1] = (_Float16)(e1 * inv);
}

// o[b,s,h,w] = sum_t att[b,h,s,t] v[b,t,h,w]; one wave does both HD tiles sharing the A fragment.
__global__ void av_kernel(const _Float16* __restrict__ atth, const _Float16* __restrict__ vT,
                          float* __restrict__ o) {
  const int lane = threadIdx.x & 31;
  const int wave = threadIdx.x >> 5;
  const int wid  = blockIdx.x * (blockDim.x >> 5) + wave;      // 32(bh)*32(mt) = 1024 waves
  const int mt = wid & 31, bh = wid >> 5;
  const int b = bh >> 3, h = bh & 7;
  const int sm = mt << 4;
  const _Float16* ar = atth + (size_t)bh * S * S;
  const _Float16* bt = vT + (size_t)b * D * S + (size_t)(h * HD) * S;
  int kend = (sm + 16 + 31) & ~31;                             // probs are 0 past row extent
  if (kend > S) kend = S;
  v8f acc0 = {}, acc1 = {};
  for (int kk = 0; kk < kend; kk += 32) {
    v16h a  = load_a_f16(ar, sm, kk, S, lane);
    acc0 = wmma32(a, load_bT_f16(bt, 0,  kk, S, lane), acc0);
    acc1 = wmma32(a, load_bT_f16(bt, 16, kk, S, lane), acc1);
  }
  const int n = lane & 15, hi = (lane >> 4) & 1;
  float* ob = o + (size_t)b * S * D + h * HD;
#pragma unroll
  for (int r = 0; r < 8; ++r) {
    ob[(size_t)(sm + r + hi * 8) * D + n]      = acc0[r];
    ob[(size_t)(sm + r + hi * 8) * D + 16 + n] = acc1[r];
  }
}

// out = LN(a + b) * s + bi, written as f32 + f16 (one 256-thread block per token)
__global__ void add_ln_kernel(const float* __restrict__ a, const float* __restrict__ b,
                              const float* __restrict__ sc, const float* __restrict__ bi,
                              float* __restrict__ out, _Float16* __restrict__ outh) {
  const int t = blockIdx.x, d = threadIdx.x;
  const float x = a[(size_t)t * D + d] + b[(size_t)t * D + d];
  __shared__ float red[256];
  red[d] = x; __syncthreads();
  for (int off = 128; off > 0; off >>= 1) { if (d < off) red[d] += red[d + off]; __syncthreads(); }
  const float mean = red[0] * (1.f / D); __syncthreads();
  const float c = x - mean;
  red[d] = c * c; __syncthreads();
  for (int off = 128; off > 0; off >>= 1) { if (d < off) red[d] += red[d + off]; __syncthreads(); }
  const float var = red[0] * (1.f / D);
  const float y = c * rsqrtf(var + EPS) * sc[d] + bi[d];
  out[(size_t)t * D + d] = y;
  outh[(size_t)t * D + d] = (_Float16)y;
}

// top-2 experts per token (argmax of logits == argmax of softmax)
__global__ void gate_kernel(const float* __restrict__ norm, const float* __restrict__ gW,
                            const float* __restrict__ gb, int* __restrict__ idx) {
  const int t = blockIdx.x, e = threadIdx.x;
  __shared__ float lg[E];
  if (e < E) {
    float acc = gb[e];
    const float* x = norm + (size_t)t * D;
    for (int d = 0; d < D; ++d) acc += x[d] * gW[(size_t)d * E + e];
    lg[e] = acc;
  }
  __syncthreads();
  if (e == 0) {
    int i0 = 0;
    for (int i = 1; i < E; ++i) if (lg[i] > lg[i0]) i0 = i;
    int i1 = (i0 == 0) ? 1 : 0;
    for (int i = 0; i < E; ++i) if (i != i0 && lg[i] > lg[i1]) i1 = i;
    idx[t * 2] = i0; idx[t * 2 + 1] = i1;
  }
}

// MoE: one wave per 16-token tile; only experts referenced by the tile are run.
// Weights pre-transposed f16: w1t/wgt [E][F][D], w2t [E][D][F].
__global__ void moe_kernel(const _Float16* __restrict__ nrmh, const int* __restrict__ idx2,
                           const _Float16* __restrict__ w1t, const float* __restrict__ f1b,
                           const _Float16* __restrict__ wgt, const float* __restrict__ fgb,
                           const _Float16* __restrict__ w2t, const float* __restrict__ f2b,
                           float* __restrict__ moe) {
  const int lane = threadIdx.x & 31;
  const int row0 = blockIdx.x << 4;
  __shared__ _Float16 hsh[16 * F];      // SiLU*gate intermediate (16 KB)
  __shared__ float    macc[16 * D];     // per-tile accumulator   (16 KB)
  __shared__ int      rexp[16][2];

  v16h af[8];                            // A fragments of norm tile, resident across experts
#pragma unroll
  for (int kk = 0; kk < 8; ++kk) af[kk] = load_a_f16(nrmh, row0, kk * 32, D, lane);

  if (lane < 16) { rexp[lane][0] = idx2[(row0 + lane) * 2]; rexp[lane][1] = idx2[(row0 + lane) * 2 + 1]; }
  for (int i = lane; i < 16 * D; i += 32) macc[i] = 0.f;
  __syncthreads();

  const int n = lane & 15, hi = (lane >> 4) & 1;
  for (int e = 0; e < E; ++e) {
    unsigned m16 = 0;
    for (int r = 0; r < 16; ++r) if (rexp[r][0] == e || rexp[r][1] == e) m16 |= 1u << r;
    if (m16 == 0) continue;                                // wave-uniform skip
    const _Float16* W1 = w1t + (size_t)e * F * D;
    const _Float16* WG = wgt + (size_t)e * F * D;
    const _Float16* W2 = w2t + (size_t)e * D * F;
    const float* b1 = f1b + (size_t)e * F;
    const float* bg = fgb + (size_t)e * F;
    const float* b2 = f2b + (size_t)e * D;
    __builtin_prefetch(W2 + lane * 64, 0, 1);              // global_prefetch while ff1 runs

    // h = silu(x@W1+b1) * (x@WG+bg) -> LDS f16   (A register-resident: 2 loads / wmma)
    for (int ft = 0; ft < F / 16; ++ft) {
      v8f a1 = {}, ag = {};
      for (int kk = 0; kk < 8; ++kk) {
        a1 = wmma32(af[kk], load_bT_f16(W1, ft * 16, kk * 32, D, lane), a1);
        ag = wmma32(af[kk], load_bT_f16(WG, ft * 16, kk * 32, D, lane), ag);
      }
      const float bb1 = b1[ft * 16 + n], bbg = bg[ft * 16 + n];
#pragma unroll
      for (int r = 0; r < 8; ++r) {
        const float hv = a1[r] + bb1;
        const float gv = ag[r] + bbg;
        const float sv = hv / (1.f + __expf(-hv));         // SiLU
        hsh[(size_t)(r + hi * 8) * F + ft * 16 + n] = (_Float16)(sv * gv);
      }
    }
    __syncthreads();

    // y = h @ W2 + b2; D blocked 4x so each LDS A fragment feeds 4 WMMAs
    for (int dtb = 0; dtb < D / 64; ++dtb) {
      v8f acc[4] = {v8f{}, v8f{}, v8f{}, v8f{}};
      for (int kk = 0; kk < 16; ++kk) {
        v16h a = load_a_f16(hsh, 0, kk * 32, F, lane);     // ds_load_b128 pair
#pragma unroll
        for (int t = 0; t < 4; ++t)
          acc[t] = wmma32(a, load_bT_f16(W2, dtb * 64 + t * 16, kk * 32, F, lane), acc[t]);
      }
#pragma unroll
      for (int t = 0; t < 4; ++t) {
        const int dn = dtb * 64 + t * 16 + n;
        const float bb2 = b2[dn];
#pragma unroll
        for (int r = 0; r < 8; ++r) {
          const int rr = r + hi * 8;
          if (m16 & (1u << rr)) macc[rr * D + dn] += acc[t][r] + bb2;
        }
      }
    }
    __syncthreads();
  }
  for (int i = lane; i < 16 * D; i += 32) moe[(size_t)row0 * D + i] = macc[i];
}

// ---------------- host ----------------
extern "C" void kernel_launch(void* const* d_in, const int* in_sizes, int n_in,
                              void* d_out, int out_size, void* d_ws, size_t ws_size,
                              hipStream_t stream) {
  const int*   tokens = (const int*)  d_in[0];
  const float* emb    = (const float*)d_in[1];
  const float* Wq = (const float*)d_in[2],  *bq = (const float*)d_in[3];
  const float* Wk = (const float*)d_in[4],  *bk = (const float*)d_in[5];
  const float* Wv = (const float*)d_in[6],  *bv = (const float*)d_in[7];
  const float* l1s = (const float*)d_in[8],  *l1b = (const float*)d_in[9];
  const float* l2s = (const float*)d_in[10], *l2b = (const float*)d_in[11];
  const float* gW = (const float*)d_in[12], *gb = (const float*)d_in[13];
  const float* f1w = (const float*)d_in[14], *f1b = (const float*)d_in[15];
  const float* fgw = (const float*)d_in[16], *fgb = (const float*)d_in[17];
  const float* f2w = (const float*)d_in[18], *f2b = (const float*)d_in[19];

  const size_t TD = (size_t)NT * D;          // 524288
  const size_t AS = (size_t)B * H * S * S;   // 8388608
  char* w = (char*)d_ws;
  auto take = [&](size_t bytes) { char* p = w; w += (bytes + 255) & ~(size_t)255; return p; };
  float*    x    = (float*)   take(TD * 4);
  float*    nrm  = (float*)   take(TD * 4);
  float*    o    = (float*)   take(TD * 4);
  float*    moe  = (float*)   take(TD * 4);
  float*    att  = (float*)   take(AS * 4);
  _Float16* xh   = (_Float16*)take(TD * 2);
  _Float16* nrmh = (_Float16*)take(TD * 2);
  _Float16* qh   = (_Float16*)take(TD * 2);
  _Float16* kh   = (_Float16*)take(TD * 2);
  _Float16* vT   = (_Float16*)take(TD * 2);
  _Float16* atth = (_Float16*)take(AS * 2);
  _Float16* wqt  = (_Float16*)take((size_t)D * D * 2);
  _Float16* wkt  = (_Float16*)take((size_t)D * D * 2);
  _Float16* wvt  = (_Float16*)take((size_t)D * D * 2);
  _Float16* w1t  = (_Float16*)take((size_t)E * D * F * 2);
  _Float16* wgt  = (_Float16*)take((size_t)E * D * F * 2);
  _Float16* w2t  = (_Float16*)take((size_t)E * F * D * 2);
  int*      idx  = (int*)     take((size_t)NT * 2 * 4);

  embed_kernel<<<(NT * D) / 256, 256, 0, stream>>>(tokens, emb, x, xh);

  const int gemm_blocks  = ((NT / 16) * (D / 64)) / 4;   // 128 (4 waves/block)
  const int score_blocks = (B * H * 32) / 4;             // 256
  const int av_blocks    = (B * H * 32) / 4;             // 256

  for (int l = 0; l < L; ++l) {
    // per-layer f16 transposed weight caches (stay L2-resident)
    convT_kernel<<<dim3(D / 32, D / 32, 1), 256, 0, stream>>>(Wq + (size_t)l * D * D, wqt, D, D, 0, 0);
    convT_kernel<<<dim3(D / 32, D / 32, 1), 256, 0, stream>>>(Wk + (size_t)l * D * D, wkt, D, D, 0, 0);
    convT_kernel<<<dim3(D / 32, D / 32, 1), 256, 0, stream>>>(Wv + (size_t)l * D * D, wvt, D, D, 0, 0);
    convT_kernel<<<dim3(D / 32, F / 32, E), 256, 0, stream>>>(f1w + (size_t)l * E * D * F, w1t, D, F,
                                                              (size_t)D * F, (size_t)F * D);
    convT_kernel<<<dim3(D / 32, F / 32, E), 256, 0, stream>>>(fgw + (size_t)l * E * D * F, wgt, D, F,
                                                              (size_t)D * F, (size_t)F * D);
    convT_kernel<<<dim3(F / 32, D / 32, E), 256, 0, stream>>>(f2w + (size_t)l * E * F * D, w2t, F, D,
                                                              (size_t)F * D, (size_t)D * F);

    gemm_qkv_kernel<<<gemm_blocks, 128, 0, stream>>>(xh, wqt, bq + (size_t)l * D, qh, 0);
    gemm_qkv_kernel<<<gemm_blocks, 128, 0, stream>>>(xh, wkt, bk + (size_t)l * D, kh, 0);
    gemm_qkv_kernel<<<gemm_blocks, 128, 0, stream>>>(xh, wvt, bv + (size_t)l * D, vT, 1);

    scores_kernel<<<score_blocks, 128, 0, stream>>>(qh, kh, att);
    softmax_kernel<<<B * H * S, 256, 0, stream>>>(att, atth);
    av_kernel<<<av_blocks, 128, 0, stream>>>(atth, vT, o);

    add_ln_kernel<<<NT, 256, 0, stream>>>(o, x, l1s + (size_t)l * D, l1b + (size_t)l * D, nrm, nrmh);

    gate_kernel<<<NT, 32, 0, stream>>>(nrm, gW + (size_t)l * D * E, gb + (size_t)l * E, idx);

    moe_kernel<<<NT / 16, 32, 0, stream>>>(nrmh, idx, w1t, f1b + (size_t)l * E * F,
                                           wgt, fgb + (size_t)l * E * F,
                                           w2t, f2b + (size_t)l * E * D, moe);

    float* dst = (l == L - 1) ? (float*)d_out : x;
    add_ln_kernel<<<NT, 256, 0, stream>>>(moe, nrm, l2s + (size_t)l * D, l2b + (size_t)l * D, dst, xh);
  }
}
